// Prompt_64415919506182
// MI455X (gfx1250) — matrix operand order repaired and checked by reading, not compiled, not run on hardware
//
#include <hip/hip_runtime.h>
#include <cstdint>
#include <cstddef>

#define B_ROWS 1024
#define D_DIM  768
#define P_ROWS 100000
#define KC     32
#define NKC    (D_DIM / KC)      // 24 k-chunks
#define LDS_STRIDE 40            // 32 bf16 + 8 pad (80 bytes) -> conflict-free b128 frag reads

typedef __attribute__((ext_vector_type(16))) __bf16 v16bf;
typedef __attribute__((ext_vector_type(8)))  float  v8f;

// branchless round-to-nearest-even f32 -> bf16 (inputs are finite; no NaN path)
__device__ __forceinline__ unsigned short f32_to_bf16_rne(float f) {
    unsigned int u = __float_as_uint(f);
    u += 0x7fffu + ((u >> 16) & 1u);
    return (unsigned short)(u >> 16);
}

// pack two f32 -> bf16x2 in one u32, branchless RNE
__device__ __forceinline__ unsigned int pack_bf16x2(float lo, float hi) {
    unsigned int ul = __float_as_uint(lo);
    unsigned int uh = __float_as_uint(hi);
    ul += 0x7fffu + ((ul >> 16) & 1u);
    uh += 0x7fffu + ((uh >> 16) & 1u);
    return (ul >> 16) | (uh & 0xffff0000u);
}

// ---------------- x_embed_norm -> bf16 ----------------
__global__ __launch_bounds__(256)
void k_xnorm(const float* __restrict__ cls, unsigned short* __restrict__ xnbf) {
    __shared__ float red[256];
    const int b = blockIdx.x, t = threadIdx.x;
    const float* row = cls + (size_t)b * D_DIM;
    float v[3]; float s = 0.f;
    for (int i = 0; i < 3; ++i) { v[i] = row[t + 256 * i]; s += v[i] * v[i]; }
    red[t] = s; __syncthreads();
    for (int off = 128; off > 0; off >>= 1) {
        if (t < off) red[t] += red[t + off];
        __syncthreads();
    }
    const float rs = rsqrtf(fmaxf(red[0], 1e-12f));
    for (int i = 0; i < 3; ++i)
        xnbf[(size_t)b * D_DIM + t + 256 * i] = f32_to_bf16_rne(v[i] * rs);
}

// ---------------- key row inverse norms (one wave32 per row) ----------------
__global__ __launch_bounds__(256)
void k_keyinv(const float* __restrict__ key, float* __restrict__ invn) {
    const int lane = threadIdx.x & 31;
    const int wave = threadIdx.x >> 5;
    const int p = blockIdx.x * 8 + wave;
    if (p >= P_ROWS) return;
    const float4* r4 = (const float4*)(key + (size_t)p * D_DIM);
    float s = 0.f;
    for (int it = 0; it < 6; ++it) {               // 6 * 32 * 4 = 768
        float4 v = r4[lane + 32 * it];
        s += v.x * v.x + v.y * v.y + v.z * v.z + v.w * v.w;
    }
    for (int off = 16; off > 0; off >>= 1) s += __shfl_xor(s, off, 32);
    if (lane == 0) invn[p] = rsqrtf(fmaxf(s, 1e-12f));
}

// ---------------- GEMM: sim[b,p] = (xn_bf16[b] . key_bf16[p]) * invn[p] ----------------
struct StageRegs { uint4 a0, a1; float4 b0, b1, b2, b3; };

__global__ __launch_bounds__(256)
void k_gemm(const unsigned short* __restrict__ xnbf,
            const float* __restrict__ key,
            const float* __restrict__ invn,
            float* __restrict__ sim) {
    __shared__ unsigned short ldsA[2][128 * LDS_STRIDE];
    __shared__ unsigned short ldsB[2][128 * LDS_STRIDE];

    const int t    = threadIdx.x;
    const int lane = t & 31;
    const int w    = t >> 5;        // wave 0..7
    const int wm   = w & 3;         // M sub-block (x32)
    const int wn   = w >> 2;        // N sub-block (x64)
    const int m0   = blockIdx.x * 128;   // M tile (8 tiles, fast grid dim -> L2 reuse of keys)
    const int n0   = blockIdx.y * 128;   // N tile over P

    const int sr = t >> 1;          // staging row 0..127
    const int sh = t & 1;           // staging half (16 elems)

    v8f acc[2][4] = {};

    union Frag { uint4 q[2]; v16bf v; };

    auto gload = [&](int kc) {
        StageRegs r;
        const uint4* asrc =
            (const uint4*)(xnbf + (size_t)(m0 + sr) * D_DIM + kc * KC + sh * 16);
        r.a0 = asrc[0]; r.a1 = asrc[1];
        int p = n0 + sr; if (p >= P_ROWS) p = P_ROWS - 1;   // clamp (cols never stored)
        const float4* bsrc =
            (const float4*)(key + (size_t)p * D_DIM + kc * KC + sh * 16);
        r.b0 = bsrc[0]; r.b1 = bsrc[1]; r.b2 = bsrc[2]; r.b3 = bsrc[3];
        return r;
    };

    auto dstore = [&](const StageRegs& r, int buf) {
        uint4* ad = (uint4*)&ldsA[buf][sr * LDS_STRIDE + sh * 16];
        ad[0] = r.a0; ad[1] = r.a1;
        const float* bf = (const float*)&r.b0;
        uint4 q0, q1;
        q0.x = pack_bf16x2(bf[0],  bf[1]);  q0.y = pack_bf16x2(bf[2],  bf[3]);
        q0.z = pack_bf16x2(bf[4],  bf[5]);  q0.w = pack_bf16x2(bf[6],  bf[7]);
        q1.x = pack_bf16x2(bf[8],  bf[9]);  q1.y = pack_bf16x2(bf[10], bf[11]);
        q1.z = pack_bf16x2(bf[12], bf[13]); q1.w = pack_bf16x2(bf[14], bf[15]);
        uint4* bd = (uint4*)&ldsB[buf][sr * LDS_STRIDE + sh * 16];
        bd[0] = q0; bd[1] = q1;
    };

    const int hl = lane >> 4;       // K-half selector per ISA 16-bit A/B layouts
    const int ll = lane & 15;

    auto compute = [&](int buf) {
        Frag a[2], b[4];
        for (int mi = 0; mi < 2; ++mi) {            // A 16x32: lane row m, K chunks hl*8 and hl*8+16
            const int m = wm * 32 + mi * 16 + ll;
            const unsigned short* base = &ldsA[buf][m * LDS_STRIDE + hl * 8];
            a[mi].q[0] = *(const uint4*)(base);
            a[mi].q[1] = *(const uint4*)(base + 16);
        }
        for (int ni = 0; ni < 4; ++ni) {            // B 32x16: lane col n, K chunks hl*8 and hl*8+16
            const int n = wn * 64 + ni * 16 + ll;
            const unsigned short* base = &ldsB[buf][n * LDS_STRIDE + hl * 8];
            b[ni].q[0] = *(const uint4*)(base);
            b[ni].q[1] = *(const uint4*)(base + 16);
        }
        for (int mi = 0; mi < 2; ++mi)
            for (int ni = 0; ni < 4; ++ni)
                acc[mi][ni] = __builtin_amdgcn_wmma_f32_16x16x32_bf16(
                    false, a[mi].v, false, b[ni].v, (short)0, acc[mi][ni], false, false);
    };

    // 3-deep software pipeline with two independent staging register sets:
    //   chunk loaded at iter kc-2 -> ds_store at iter kc -> WMMA at iter kc+1.
    StageRegs rE = gload(0);        // even chunks
    dstore(rE, 0);
    StageRegs rO = gload(1);        // odd chunks
    rE = gload(2);
    __syncthreads();

    for (int kc = 0; kc < NKC; kc += 2) {
        // compute even buffer; stage odd chunk kc+1; prefetch chunk kc+3
        if (kc + 1 < NKC) {
            dstore(rO, 1);
            if (kc + 3 < NKC) rO = gload(kc + 3);
        }
        compute(0);
        __syncthreads();
        if (kc + 1 >= NKC) break;
        // compute odd buffer; stage even chunk kc+2; prefetch chunk kc+4
        if (kc + 2 < NKC) {
            dstore(rE, 0);
            if (kc + 4 < NKC) rE = gload(kc + 4);
        }
        compute(1);
        __syncthreads();
    }

    // epilogue: scale columns by invnorm, non-temporal store (keep key tiles resident in L2)
    for (int ni = 0; ni < 4; ++ni) {
        const int p = n0 + wn * 64 + ni * 16 + ll;
        if (p >= P_ROWS) continue;
        const float s = invn[p];
        for (int mi = 0; mi < 2; ++mi) {
            const int mbase = m0 + wm * 32 + mi * 16 + hl * 8;
            for (int vi = 0; vi < 8; ++vi)
                __builtin_nontemporal_store(acc[mi][ni][vi] * s,
                                            &sim[(size_t)(mbase + vi) * P_ROWS + p]);
        }
    }
}

// ---------------- per-row top-2 over 100000 columns ----------------
__global__ __launch_bounds__(256)
void k_top2(const float* __restrict__ sim, int* __restrict__ idx_ws,
            float* __restrict__ rowsum, int* __restrict__ idx_out) {
    __shared__ float sv0[256], sv1[256];
    __shared__ int   si0[256], si1[256];
    const int b = blockIdx.x, t = threadIdx.x;
    const float* row = sim + (size_t)b * P_ROWS;
    float v0 = -INFINITY, v1 = -INFINITY;
    int   i0 = 0x7fffffff, i1 = 0x7fffffff;
    for (int p = t; p < P_ROWS; p += 256) {
        const float x = __builtin_nontemporal_load(row + p);
        if (x > v0 || (x == v0 && p < i0)) { v1 = v0; i1 = i0; v0 = x; i0 = p; }
        else if (x > v1 || (x == v1 && p < i1)) { v1 = x; i1 = p; }
    }
    sv0[t] = v0; sv1[t] = v1; si0[t] = i0; si1[t] = i1;
    __syncthreads();
    for (int off = 128; off > 0; off >>= 1) {
        if (t < off) {
            const float w0 = sv0[t + off], w1 = sv1[t + off];
            const int   j0 = si0[t + off], j1 = si1[t + off];
            if (w0 > v0 || (w0 == v0 && j0 < i0)) { v1 = v0; i1 = i0; v0 = w0; i0 = j0; }
            else if (w0 > v1 || (w0 == v1 && j0 < i1)) { v1 = w0; i1 = j0; }
            if (w1 > v1 || (w1 == v1 && j1 < i1)) { v1 = w1; i1 = j1; }
            sv0[t] = v0; sv1[t] = v1; si0[t] = i0; si1[t] = i1;
        }
        __syncthreads();
    }
    if (t == 0) {
        idx_ws[b * 2] = i0; idx_ws[b * 2 + 1] = i1;
        rowsum[b] = v0 + v1;
        idx_out[b * 2] = i0; idx_out[b * 2 + 1] = i1;
    }
}

// ---------------- batched_prompt = mean(prompt[idx]) + cls ----------------
__global__ __launch_bounds__(256)
void k_bp(const float* __restrict__ prompt, const float* __restrict__ cls,
          const int* __restrict__ idx_ws, float* __restrict__ outp) {
    const int b = blockIdx.x, t = threadIdx.x;
    const int i0 = idx_ws[b * 2], i1 = idx_ws[b * 2 + 1];
    const float* p0 = prompt + (size_t)i0 * D_DIM;
    const float* p1 = prompt + (size_t)i1 * D_DIM;
    const float* c  = cls + (size_t)b * D_DIM;
    float* o = outp + (size_t)b * D_DIM;
    for (int j = t; j < D_DIM; j += 256)
        o[j] = 0.5f * (p0[j] + p1[j]) + c[j];
}

// ---------------- deterministic reduce_sim ----------------
__global__ __launch_bounds__(256)
void k_rsum(const float* __restrict__ rowsum, float* __restrict__ out) {
    __shared__ float red[256];
    const int t = threadIdx.x;
    float s = 0.f;
    for (int i = t; i < B_ROWS; i += 256) s += rowsum[i];
    red[t] = s; __syncthreads();
    for (int off = 128; off > 0; off >>= 1) {
        if (t < off) red[t] += red[t + off];
        __syncthreads();
    }
    if (t == 0) out[0] = red[0] / (float)B_ROWS;
}

extern "C" void kernel_launch(void* const* d_in, const int* in_sizes, int n_in,
                              void* d_out, int out_size, void* d_ws, size_t ws_size,
                              hipStream_t stream) {
    const float* cls    = (const float*)d_in[0];   // [1024,768]
    const float* prompt = (const float*)d_in[1];   // [100000,768]
    const float* pkey   = (const float*)d_in[2];   // [100000,768]
    float* out = (float*)d_out;

    // workspace layout (~2 MB)
    char* ws = (char*)d_ws;
    unsigned short* xnbf = (unsigned short*)ws;                       // 1024*768*2 = 1,572,864 B
    float* invn   = (float*)(ws + 1572864);                           // 100000*4   =   400,000 B
    int*   idx_ws = (int*)  (ws + 1572864 + 400000);                  //               8,192 B
    float* rowsum = (float*)(ws + 1572864 + 400000 + 8192);           //               4,096 B

    // output layout: batched_prompt [1024*768] | reduce_sim [1] | similarity [1024*100000] | idx [2048]
    float* bp_out   = out;
    float* rsum_out = out + (size_t)B_ROWS * D_DIM;
    float* sim      = out + (size_t)B_ROWS * D_DIM + 1;
    int*   idx_out  = (int*)(out + (size_t)B_ROWS * D_DIM + 1 + (size_t)B_ROWS * P_ROWS);

    k_xnorm <<<B_ROWS, 256, 0, stream>>>(cls, xnbf);
    k_keyinv<<<(P_ROWS + 7) / 8, 256, 0, stream>>>(pkey, invn);
    dim3 g(8, (P_ROWS + 127) / 128);   // M fastest -> key tiles shared in L2 across the 8 M-tiles
    k_gemm  <<<g, 256, 0, stream>>>(xnbf, pkey, invn, sim);
    k_top2  <<<B_ROWS, 256, 0, stream>>>(sim, idx_ws, rowsum, idx_out);
    k_bp    <<<B_ROWS, 256, 0, stream>>>(prompt, cls, idx_ws, bp_out);
    k_rsum  <<<1, 256, 0, stream>>>(rowsum, rsum_out);
}
